// Attention_36601711297049
// MI455X (gfx1250) — matrix operand order repaired and checked by reading, not compiled, not run on hardware
//
#include <hip/hip_runtime.h>

typedef __attribute__((ext_vector_type(16))) __bf16 v16bf;
typedef __attribute__((ext_vector_type(8)))  __bf16 v8bf;
typedef __attribute__((ext_vector_type(4)))  __bf16 v4bf;
typedef __attribute__((ext_vector_type(8)))  float  v8f;

union bfrag { v16bf v; v8bf h[2]; };

#define HEADS  4
#define DIMH   32
#define C_IN   256
#define HIDDEN 128
#define HW     4096
#define NB     2

// Fragment storage: each 16x32 (bf16) fragment = 32 lanes x 16 elems = 512 bf16,
// stored lane-major so a lane loads its 16 contiguous elements (32B) directly.
#define FRAG_ELEMS   512
#define BH_STRIDE    131072        // 256 frags * 512 per (b,head)
#define OF_B_STRIDE  524288        // 4 kb-blocks * 256 i-tiles * 512
#define XSTR         264           // LDS row stride (256 + 8 pad) in bf16

static __device__ __forceinline__ v8f wmma_bf16(v16bf a, v16bf b, v8f c) {
    return __builtin_amdgcn_wmma_f32_16x16x32_bf16(
        false, a, false, b, (short)0, c, false, false);
}

// hardware exp2 (v_exp_f32)
static __device__ __forceinline__ float exp2_hw(float x) {
    return __builtin_amdgcn_exp2f(x);
}

// -------------------------------------------------------------------------
// Kernel A: qkv = W_qkv @ X, written as pre-swizzled bf16 WMMA fragments.
// Tiles are reordered so each 16-row tile has uniform (s, head, dhalf):
//   s=0 (Q): W x X tile  -> B-frag (K-dim = d), scaled by log2e/sqrt(32)
//   s=1 (K): W x X tile  -> A-frag (rows = j, K-dim = d)
//   s=2 (V): X^T x Wv^T  -> A-frag (rows = d, K-dim = j)  [computed transposed
//            so the D layout coincides with the fragment layout]
// X tile staged TRANSPOSED in LDS ([i][c], padded) so every fragment is built
// from two 16-byte ds_load_b128 per lane. One packed 16B store per tile/lane.
// -------------------------------------------------------------------------
__global__ __launch_bounds__(128)
void qkv_kernel(const float* __restrict__ x, const float* __restrict__ wqkv,
                __bf16* __restrict__ qf, __bf16* __restrict__ kf,
                __bf16* __restrict__ vf)
{
    __shared__ __bf16 xs[16 * XSTR];   // X tile transposed: 16 positions x 256 ch
    const int b  = blockIdx.y;
    const int it = blockIdx.x;         // i-tile 0..255
    const int i0 = it * 16;
    const int tid = threadIdx.x;

    // stage: thread (ti, tc) loads 4 channel-strided floats at position ti,
    // stores one packed 8-byte v4bf row segment.
    const float* xb = x + (size_t)b * C_IN * HW;
    {
        const int ti = tid & 15;          // position within tile
        const int tc = (tid >> 4) * 4;    // channel group base (0..28, step 4)
        for (int k = 0; k < 8; ++k) {
            int c0 = tc + k * 32;
            v4bf pk;
#pragma unroll
            for (int j = 0; j < 4; ++j)
                pk[j] = (__bf16)xb[(size_t)(c0 + j) * HW + i0 + ti];
            *(v4bf*)(xs + ti * XSTR + c0) = pk;
        }
    }
    __syncthreads();

    const int lane   = tid & 31;
    const int wave   = tid >> 5;
    const int laneHi = lane >> 4;
    const int lmod   = lane & 15;
    const __bf16* xrow = xs + lmod * XSTR;     // row i (or column of X)
    const float qscale = 0.2550181707556564f;  // (1/sqrt(32)) * log2(e)

    for (int mt = wave; mt < 24; mt += 4) {
        const int s     = mt >> 3;        // 0=q 1=k 2=v   (wave-uniform)
        const int head  = (mt >> 1) & 3;  // wave-uniform
        const int dhalf = mt & 1;         // wave-uniform
        const int d0    = dhalf * 16;
        const size_t bh = (size_t)(b * HEADS + head) * BH_STRIDE;
        // physical W row for logical (head, d = d0+lmod, s); stride-3 interleave
        const float* wrow = wqkv + (size_t)(head * 96 + (d0 + lmod) * 3 + s) * C_IN;

        v8f acc = {0.f,0.f,0.f,0.f,0.f,0.f,0.f,0.f};

        if (s < 2) {
            // ---- tile: rows = 16 channels (d), cols = 16 positions (i) ----
            for (int kb = 0; kb < 8; ++kb) {
                bfrag a, bx;
#pragma unroll
                for (int e = 0; e < 16; ++e) {   // A-frag of W (gathered rows)
                    int k = kb * 32 + e + ((e & 8) ? 8 : 0) + (laneHi ? 8 : 0);
                    a.v[e] = (__bf16)wrow[k];
                }
                // B-frag of X: row lmod, two 8-elem runs at c = kb*32 + 8*laneHi (+16)
                bx.h[0] = *(const v8bf*)(xrow + kb * 32 + laneHi * 8);
                bx.h[1] = *(const v8bf*)(xrow + kb * 32 + laneHi * 8 + 16);
                acc = wmma_bf16(a.v, bx.v, acc);
            }
            v8bf pk;
            if (s == 0) {
#pragma unroll
                for (int r = 0; r < 8; ++r) pk[r] = (__bf16)(acc[r] * qscale);
                // Q B-frag: lane' = lane, elems r + 8*dhalf
                *(v8bf*)(qf + bh + (size_t)it * FRAG_ELEMS
                              + lane * 16 + 8 * dhalf) = pk;
            } else {
#pragma unroll
                for (int r = 0; r < 8; ++r) pk[r] = (__bf16)acc[r];
                // K A-frag: lane' = lane, elems r + 8*dhalf
                *(v8bf*)(kf + bh + (size_t)it * FRAG_ELEMS
                              + lane * 16 + 8 * dhalf) = pk;
            }
        } else {
            // ---- V, transposed tile: rows = 16 positions (i), cols = 16 d ----
            for (int kb = 0; kb < 8; ++kb) {
                bfrag a, bw;
                // A-frag of X^T: row lmod, kmap = two contiguous 8-elem runs
                a.h[0] = *(const v8bf*)(xrow + kb * 32 + 8 * laneHi);
                a.h[1] = *(const v8bf*)(xrow + kb * 32 + 8 * laneHi + 16);
#pragma unroll
                for (int e = 0; e < 16; ++e)     // B-frag of Wv^T (two 8-elem runs)
                    bw.v[e] = (__bf16)wrow[kb * 32 + e + ((e & 8) ? 8 : 0) + (laneHi ? 8 : 0)];
                acc = wmma_bf16(a.v, bw.v, acc);
            }
            v8bf pk;
#pragma unroll
            for (int r = 0; r < 8; ++r) pk[r] = (__bf16)acc[r];
            // V A-frag (jb = it>>1, dh = dhalf): lane' = lane, elems r + 8*(it&1)
            *(v8bf*)(vf + bh + (size_t)((it >> 1) * 2 + dhalf) * FRAG_ELEMS
                          + lane * 16 + 8 * (it & 1)) = pk;
        }
    }
}

// -------------------------------------------------------------------------
// Kernel B: flash attention. One wave per (b, head, 16-query tile).
//   sim^T tile = WMMA(A = K^T frag, B = Q frag)  -> lane holds 8 j's of one i
//   online softmax in base-2 (scale carries log2e), one shfl_xor(16) per
//   reduction; P^T B-frag built from each lane's own p values; out^T accumulated
//   with WMMA(A = V frag, B = P^T frag).
// -------------------------------------------------------------------------
__global__ __launch_bounds__(128)
void attn_kernel(const __bf16* __restrict__ qf, const __bf16* __restrict__ kf,
                 const __bf16* __restrict__ vf, __bf16* __restrict__ of)
{
    const int wid  = blockIdx.x * 4 + (threadIdx.x >> 5);   // 0..2047
    const int lane = threadIdx.x & 31;
    const int it   = wid & 255;
    const int bh   = wid >> 8;           // b*4 + head
    const int b    = bh >> 2;
    const int head = bh & 3;
    const int laneHi = lane >> 4;

    const __bf16* kbase = kf + (size_t)bh * BH_STRIDE;
    const __bf16* vbase = vf + (size_t)bh * BH_STRIDE;
    v16bf qfrag = *(const v16bf*)(qf + (size_t)bh * BH_STRIDE
                                     + (size_t)it * FRAG_ELEMS + lane * 16);

    v8f acc0 = {0.f,0.f,0.f,0.f,0.f,0.f,0.f,0.f};
    v8f acc1 = {0.f,0.f,0.f,0.f,0.f,0.f,0.f,0.f};
    float m_run = -1e30f, l_run = 0.f;

    for (int jb = 0; jb < 128; ++jb) {           // keys in blocks of 32
        v16bf k0 = *(const v16bf*)(kbase + (size_t)(2 * jb)     * FRAG_ELEMS + lane * 16);
        v16bf k1 = *(const v16bf*)(kbase + (size_t)(2 * jb + 1) * FRAG_ELEMS + lane * 16);
        v8f z = {0.f,0.f,0.f,0.f,0.f,0.f,0.f,0.f};
        v8f s0 = wmma_bf16(k0, qfrag, z);
        v8f s1 = wmma_bf16(k1, qfrag, z);

        if (jb < 127) {   // prefetch next K/V fragments (global_prefetch_b8)
            __builtin_prefetch(kbase + (size_t)(2 * jb + 2) * FRAG_ELEMS + lane * 16, 0, 3);
            __builtin_prefetch(vbase + (size_t)(2 * jb + 2) * FRAG_ELEMS + lane * 16, 0, 3);
        }

        // ---- online softmax (base-2) over this 32-key block ----
        float mt = -1e30f;
#pragma unroll
        for (int r = 0; r < 8; ++r) { mt = fmaxf(mt, s0[r]); mt = fmaxf(mt, s1[r]); }
        mt = fmaxf(mt, __shfl_xor(mt, 16, 32));
        float m_new = fmaxf(m_run, mt);
        float corr  = exp2_hw(m_run - m_new);

        float p0[8], p1[8];
        float psum = 0.f;
#pragma unroll
        for (int r = 0; r < 8; ++r) {
            p0[r] = exp2_hw(s0[r] - m_new);
            p1[r] = exp2_hw(s1[r] - m_new);
            psum += p0[r] + p1[r];
        }
        psum += __shfl_xor(psum, 16, 32);
        l_run = l_run * corr + psum;
        m_run = m_new;
#pragma unroll
        for (int r = 0; r < 8; ++r) { acc0[r] *= corr; acc1[r] *= corr; }

        // ---- build P^T B-fragment (32 j-rows x 16 i-cols) ----
        v16bf bp;
#pragma unroll
        for (int r = 0; r < 8; ++r) {
            float o0 = __shfl_xor(p0[r], 16, 32);
            float o1 = __shfl_xor(p1[r], 16, 32);
            float lo = p0[r];   // K rows 8*laneHi + r
            float hi = p1[r];   // K rows 16 + 8*laneHi + r
            bp[r]     = (__bf16)lo;
            bp[r + 8] = (__bf16)hi;
        }

        v16bf v0 = *(const v16bf*)(vbase + (size_t)(2 * jb)     * FRAG_ELEMS + lane * 16);
        v16bf v1 = *(const v16bf*)(vbase + (size_t)(2 * jb + 1) * FRAG_ELEMS + lane * 16);
        acc0 = wmma_bf16(v0, bp, acc0);
        acc1 = wmma_bf16(v1, bp, acc1);
    }

    // epilogue: out[i][d] = acc/l, packed B-fragment stores (K-dim = c)
    float inv_l = 1.0f / l_run;
    __bf16* ob = of + (size_t)b * OF_B_STRIDE;
    size_t fbase = (size_t)(head * 256 + it) * FRAG_ELEMS;
    int icol = lane & 15;
    v8bf pk0, pk1;
#pragma unroll
    for (int r = 0; r < 8; ++r) {
        pk0[r] = (__bf16)(acc0[r] * inv_l);     // d = r + 8*laneHi      (< 16)
        pk1[r] = (__bf16)(acc1[r] * inv_l);     // d = 16 + r + 8*laneHi
    }
    *(v8bf*)(ob + fbase + icol * 16 + 8 * laneHi)        = pk0;
    *(v8bf*)(ob + fbase + (icol + 16) * 16 + 8 * laneHi) = pk1;
}

// -------------------------------------------------------------------------
// Kernel C: y = W_out @ O + b_out.  One wave per 16x16 output tile, K=128.
// -------------------------------------------------------------------------
__global__ __launch_bounds__(128)
void proj_kernel(const __bf16* __restrict__ of, const float* __restrict__ wout,
                 const float* __restrict__ bout, float* __restrict__ y)
{
    const int wid  = blockIdx.x * 4 + (threadIdx.x >> 5);   // 0..8191
    const int lane = threadIdx.x & 31;
    const int it   = wid & 255;
    const int mt   = (wid >> 8) & 15;
    const int b    = wid >> 12;
    const int laneHi = lane >> 4;
    const int lmod   = lane & 15;

    const __bf16* ob = of + (size_t)b * OF_B_STRIDE;
    const float* wrow = wout + (size_t)(mt * 16 + lmod) * HIDDEN;
    v8f acc = {0.f,0.f,0.f,0.f,0.f,0.f,0.f,0.f};
    for (int kb = 0; kb < 4; ++kb) {
        v16bf a;
#pragma unroll
        for (int e = 0; e < 16; ++e) {
            int k = kb * 32 + laneHi * 16 + e;
            a[e] = (__bf16)wrow[k];
        }
        v16bf bx = *(const v16bf*)(ob + (size_t)(kb * 256 + it) * FRAG_ELEMS + lane * 16);
        acc = wmma_bf16(a, bx, acc);
    }
    float* yb = y + (size_t)b * C_IN * HW;
#pragma unroll
    for (int r = 0; r < 8; ++r) {
        int o = mt * 16 + r + 8 * laneHi;
        yb[(size_t)o * HW + it * 16 + lmod] = acc[r] + bout[o];
    }
}

// -------------------------------------------------------------------------
extern "C" void kernel_launch(void* const* d_in, const int* in_sizes, int n_in,
                              void* d_out, int out_size, void* d_ws, size_t ws_size,
                              hipStream_t stream)
{
    const float* x    = (const float*)d_in[0];   // (2,256,64,64)
    const float* wqkv = (const float*)d_in[1];   // (384,256)
    const float* wout = (const float*)d_in[2];   // (256,128)
    const float* bout = (const float*)d_in[3];   // (256,)
    float* y = (float*)d_out;

    __bf16* ws = (__bf16*)d_ws;                  // 8 MB of bf16 fragments
    __bf16* qf = ws;                             // 2*4*131072
    __bf16* kf = ws + 1048576;
    __bf16* vf = ws + 2097152;
    __bf16* of = ws + 3145728;

    qkv_kernel<<<dim3(256, 2), 128, 0, stream>>>(x, wqkv, qf, kf, vf);
    attn_kernel<<<512, 128, 0, stream>>>(qf, kf, vf, of);
    proj_kernel<<<2048, 128, 0, stream>>>(of, wout, bout, y);
}
